// DecoderBlock_89455578841310
// MI455X (gfx1250) — matrix operand order repaired and checked by reading, not compiled, not run on hardware
//
#include <hip/hip_runtime.h>
#include <hip/hip_bf16.h>

// Decoder block for MI455X (gfx1250): f16 WMMA (16x16x32) everywhere,
// fp32 accumulation, zero-padded head dim 21->32, Wo rows 126->128.
// Weights pre-swizzled into WMMA B-fragment order (32B contiguous per lane).
// Large global->LDS tile staging done by the Tensor Data Mover (TDM),
// waited with s_wait_tensorcnt.

#define B_   512
#define T_   128
#define D_   128
#define H_   6
#define DH   21
#define DHP  32
#define DFF  512
#define EPS_ 1e-5f

typedef __attribute__((ext_vector_type(16))) _Float16 v16h;
typedef __attribute__((ext_vector_type(8)))  float    v8f;
typedef __attribute__((ext_vector_type(4)))  unsigned int u32x4;
typedef __attribute__((ext_vector_type(8)))  int          i32x8;
typedef __attribute__((ext_vector_type(4)))  int          i32x4;
typedef _Float16 half_t;

extern __shared__ __align__(16) char smem_raw[];

// ---------------- TDM: 1-D global -> LDS DMA (wave-level) -------------------
// D# per CDNA5 ISA. data_size=8B, tile_dim0 = n8 (<=65535),
// tile_dim1/2 = 0 (1-D), groups 2/3 zero. Issue from one wave, then
// s_wait_tensorcnt 0 and barrier.
static __device__ __forceinline__ void tdm_load_1d(unsigned lds_off,
                                                   const void* gsrc,
                                                   unsigned n8 /* 8-byte units */) {
  const unsigned long long ga = (unsigned long long)(size_t)gsrc;
  u32x4 g0;
  g0[0] = 1u;                                        // count=1 (valid user D#)
  g0[1] = lds_off;                                   // lds_addr (bytes)
  g0[2] = (unsigned)(ga & 0xffffffffu);              // global_addr[31:0]
  g0[3] = (unsigned)((ga >> 32) & 0x01ffffffu) | (2u << 30);  // addr[56:32]|type=2
  i32x8 g1;
  g1[0] = 0x00030000;                  // workgroup_mask=0, data_size=3 (8B)
  g1[1] = (int)(n8 << 16);             // tensor_dim0[15:0] in bits[63:48]
  g1[2] = (int)((n8 >> 16) | (1u << 16));  // tensor_dim0[31:16] | tensor_dim1=1
  g1[3] = (int)(n8 << 16);             // tensor_dim1 hi=0 | tile_dim0 = n8
  g1[4] = 0;                           // tile_dim1=0 (1-D), tile_dim2=0
  g1[5] = (int)n8;                     // tensor_dim0_stride low
  g1[6] = 0;
  g1[7] = 0;
  const i32x4 z4 = {0, 0, 0, 0};
#if __clang_major__ >= 23
  const i32x8 z8 = {0, 0, 0, 0, 0, 0, 0, 0};
  __builtin_amdgcn_tensor_load_to_lds(g0, g1, z4, z4, z8, 0);
#else
  __builtin_amdgcn_tensor_load_to_lds(g0, g1, z4, z4, 0);
#endif
}

// ---------------- WMMA fragment helpers (wave32, 16-bit layouts) ------------

// A matrix 16x32 (MxK), row-major source with element stride ld.
static __device__ __forceinline__ v16h frag_A(const half_t* p, int ld) {
  const int lane = threadIdx.x & 31;
  const int m = lane & 15, hs = lane >> 4;
  v16h a;
#pragma unroll
  for (int i = 0; i < 16; ++i) {
    const int k = i + hs * 16;
    a[i] = p[m * ld + k];
  }
  return a;
}

// B matrix 32x16 (KxN), row-major source p[k*ld + n].
static __device__ __forceinline__ v16h frag_B(const half_t* p, int ld) {
  const int lane = threadIdx.x & 31;
  const int n = lane & 15, hs = lane >> 4;
  v16h b;
#pragma unroll
  for (int i = 0; i < 16; ++i) b[i] = p[(i + 16 * hs) * ld + n];
  return b;
}

// B fragment from pre-swizzled storage: lane's 16 halves are contiguous.
static __device__ __forceinline__ v16h frag_B_packed(const half_t* p) {
  return *(reinterpret_cast<const v16h*>(p) + (threadIdx.x & 31));
}

// B matrix 32x16 where the source is stored N-major (transposed read).
static __device__ __forceinline__ v16h frag_BT(const half_t* p, int ld) {
  const int lane = threadIdx.x & 31;
  const int n = lane & 15, hs = lane >> 4;
  v16h b;
#pragma unroll
  for (int i = 0; i < 16; ++i) b[i] = p[n * ld + i + 16 * hs];
  return b;
}

static __device__ __forceinline__ v8f wmma16(v16h a, v16h b, v8f c) {
  return __builtin_amdgcn_wmma_f32_16x16x32_f16(false, a, false, b,
                                                (short)0, c, false, false);
}

static __device__ __forceinline__ float wave_sum(float v) {
#pragma unroll
  for (int o = 16; o > 0; o >>= 1) v += __shfl_xor(v, o, 32);
  return v;
}
static __device__ __forceinline__ float wave_max(float v) {
#pragma unroll
  for (int o = 16; o > 0; o >>= 1) v = fmaxf(v, __shfl_xor(v, o, 32));
  return v;
}

// ---------------- K0: pack weights to padded f16, B-fragment order ---------

static __device__ __forceinline__ void frag_kn(int j, int numK, int* k, int* n) {
  const int tile = j >> 9;
  const int lane = (j >> 4) & 31;
  const int i = j & 15;
  *k = (tile % numK) * 32 + i + 16 * (lane >> 4);
  *n = (tile / numK) * 16 + (lane & 15);
}

__global__ __launch_bounds__(256) void k_pack(
    const float* __restrict__ Wq, const float* __restrict__ Wk,
    const float* __restrict__ Wv, const float* __restrict__ Wo,
    const float* __restrict__ W1, const float* __restrict__ W2,
    half_t* __restrict__ WqkvP, half_t* __restrict__ WoP,
    half_t* __restrict__ W1P, half_t* __restrict__ W2P) {
  const int nQKV = H_ * 3 * D_ * DHP;  // 73728
  const int nWo  = D_ * D_;            // 16384
  const int nW1  = D_ * DFF;           // 65536
  const int nW2  = DFF * D_;           // 65536
  int i = blockIdx.x * 256 + threadIdx.x;
  int k, n;
  if (i < nQKV) {
    const int j = i & 4095;
    const int hm = i >> 12;
    const int mat = hm % 3, h = hm / 3;
    frag_kn(j, 4, &k, &n);
    const float* src = (mat == 0) ? Wq : (mat == 1) ? Wk : Wv;
    WqkvP[i] = (half_t)((n < DH) ? src[(h * D_ + k) * DH + n] : 0.0f);
  } else if (i < nQKV + nWo) {
    const int j = i - nQKV;
    frag_kn(j, 4, &k, &n);
    WoP[j] = (half_t)((k < 126) ? Wo[k * D_ + n] : 0.0f);
  } else if (i < nQKV + nWo + nW1) {
    const int j = i - nQKV - nWo;
    frag_kn(j, 4, &k, &n);
    W1P[j] = (half_t)W1[k * DFF + n];
  } else if (i < nQKV + nWo + nW1 + nW2) {
    const int j = i - nQKV - nWo - nW1;
    frag_kn(j, 16, &k, &n);
    W2P[j] = (half_t)W2[k * D_ + n];
  }
}

// ---------------- K1: LayerNorm1 -> f16, one wave per row ------------------

__global__ __launch_bounds__(256) void k_ln1(
    const float* __restrict__ x, const float* __restrict__ g,
    const float* __restrict__ be, half_t* __restrict__ out) {
  const int row = blockIdx.x * 8 + (threadIdx.x >> 5);
  const int lane = threadIdx.x & 31;
  const float4 v = *reinterpret_cast<const float4*>(x + (size_t)row * D_ + lane * 4);
  float s  = v.x + v.y + v.z + v.w;
  float s2 = v.x * v.x + v.y * v.y + v.z * v.z + v.w * v.w;
  s = wave_sum(s);
  s2 = wave_sum(s2);
  const float mu = s * (1.0f / D_);
  const float var = s2 * (1.0f / D_) - mu * mu;
  const float rs = rsqrtf(var + EPS_);
  const float4 gg = *reinterpret_cast<const float4*>(g + lane * 4);
  const float4 bb = *reinterpret_cast<const float4*>(be + lane * 4);
  half_t* o = out + (size_t)row * D_ + lane * 4;
  o[0] = (half_t)((v.x - mu) * rs * gg.x + bb.x);
  o[1] = (half_t)((v.y - mu) * rs * gg.y + bb.y);
  o[2] = (half_t)((v.z - mu) * rs * gg.z + bb.z);
  o[3] = (half_t)((v.w - mu) * rs * gg.w + bb.w);
}

// ---------------- K2: fused QKV + causal attention, one block per (b,h) ----

__global__ __launch_bounds__(256) void k_attn(
    const half_t* __restrict__ hF, const half_t* __restrict__ Wqkv,
    half_t* __restrict__ oF) {
  half_t* h_s = (half_t*)smem_raw;                  // 128x128 f16; later probs
  half_t* q_s = (half_t*)(smem_raw + 32768);        // 128x32
  half_t* k_s = (half_t*)(smem_raw + 40960);        // 128x32
  half_t* v_s = (half_t*)(smem_raw + 49152);        // 128x32
  float*  s_s = (float*)(smem_raw + 57344);         // 128x128 f32

  const int bb = blockIdx.x / H_;
  const int hh = blockIdx.x % H_;
  const int tid = threadIdx.x;
  const int wave = tid >> 5;
  const int lane = tid & 31;
  const int fn = lane & 15, fhs = lane >> 4;

  // TDM: DMA the h tile (32KB contiguous) into LDS; wave 0 issues & waits.
  if (wave == 0) {
    tdm_load_1d(0u, hF + (size_t)bb * T_ * D_, (T_ * D_ * 2) / 8);
    __builtin_amdgcn_s_wait_tensorcnt(0);
  }
  __syncthreads();

  // q,k,v = h @ W{q,k,v}[h]  (padded N=32): 48 tiles over 8 waves
  const half_t* Wbase = Wqkv + (size_t)hh * 3 * D_ * DHP;
  for (int t = wave; t < 48; t += 8) {
    const int mat = t >> 4, rem = t & 15, mt = rem >> 1, nt = rem & 1;
    const half_t* W = Wbase + mat * D_ * DHP;  // packed, numK=4
    v8f c = {};
#pragma unroll
    for (int ks = 0; ks < 4; ++ks) {
      v16h a = frag_A(h_s + mt * 16 * D_ + ks * 32, D_);
      v16h b = frag_B_packed(W + ((nt * 4 + ks) << 9));
      c = wmma16(a, b, c);
    }
    half_t* dstm = (mat == 0) ? q_s : (mat == 1) ? k_s : v_s;
    const float sc = (mat == 0) ? 0.21821789023599236f : 1.0f;  // 1/sqrt(21)
#pragma unroll
    for (int r = 0; r < 8; ++r) {
      const int m = r + 8 * fhs;
      dstm[(mt * 16 + m) * DHP + nt * 16 + fn] = (half_t)(c[r] * sc);
    }
  }
  __syncthreads();

  // scores: wave owns M-tile row `wave` (rows 16w..16w+15), full K=32
  {
    v16h a = frag_A(q_s + wave * 16 * DHP, DHP);
#pragma unroll
    for (int nt = 0; nt < 8; ++nt) {
      v16h b = frag_BT(k_s + nt * 16 * DHP, DHP);  // k^T
      v8f c = {};
      c = wmma16(a, b, c);
#pragma unroll
      for (int r = 0; r < 8; ++r) {
        const int m = r + 8 * fhs;
        s_s[(wave * 16 + m) * T_ + nt * 16 + fn] = c[r];
      }
    }
  }

  // causal softmax on own 16 rows; probs (f16) overwrite h_s
  for (int r = 0; r < 16; ++r) {
    const int row = wave * 16 + r;
    const int c0 = lane * 4;
    const float4 sv = *reinterpret_cast<const float4*>(s_s + row * T_ + c0);
    const float NEG = -__builtin_inff();
    float e0 = (c0 + 0 <= row) ? sv.x : NEG;
    float e1 = (c0 + 1 <= row) ? sv.y : NEG;
    float e2 = (c0 + 2 <= row) ? sv.z : NEG;
    float e3 = (c0 + 3 <= row) ? sv.w : NEG;
    float mx = wave_max(fmaxf(fmaxf(e0, e1), fmaxf(e2, e3)));
    float p0 = (c0 + 0 <= row) ? __expf(sv.x - mx) : 0.0f;
    float p1 = (c0 + 1 <= row) ? __expf(sv.y - mx) : 0.0f;
    float p2 = (c0 + 2 <= row) ? __expf(sv.z - mx) : 0.0f;
    float p3 = (c0 + 3 <= row) ? __expf(sv.w - mx) : 0.0f;
    const float inv = 1.0f / wave_sum(p0 + p1 + p2 + p3);
    half_t* pr = h_s + row * T_ + c0;
    pr[0] = (half_t)(p0 * inv);
    pr[1] = (half_t)(p1 * inv);
    pr[2] = (half_t)(p2 * inv);
    pr[3] = (half_t)(p3 * inv);
  }

  // o = probs @ v : wave's own M rows; write cols < 21 into concat buffer
#pragma unroll
  for (int nt = 0; nt < 2; ++nt) {
    v8f c = {};
#pragma unroll
    for (int ks = 0; ks < 4; ++ks) {
      v16h a = frag_A(h_s + wave * 16 * T_ + ks * 32, T_);
      v16h b = frag_B(v_s + ks * 32 * DHP + nt * 16, DHP);
      c = wmma16(a, b, c);
    }
#pragma unroll
    for (int r = 0; r < 8; ++r) {
      const int m = r + 8 * fhs;
      const int col = nt * 16 + fn;
      if (col < DH)
        oF[((size_t)bb * T_ + wave * 16 + m) * D_ + hh * DH + col] = (half_t)c[r];
    }
  }
}

// ---------------- K3: x1 = x + o@Wo + bo -> d_out ; fused LN2 -> h2 --------

__global__ __launch_bounds__(256) void k_proj_ln2(
    const half_t* __restrict__ oF, const half_t* __restrict__ WoP,
    const float* __restrict__ x, const float* __restrict__ bo,
    const float* __restrict__ g2, const float* __restrict__ be2,
    float* __restrict__ out, half_t* __restrict__ h2F) {
  half_t* o_s  = (half_t*)smem_raw;            // 128x128 f16 (32KB)
  float*  x1_s = (float*)(smem_raw + 32768);   // 128x128 f32 (64KB)
  const int row0 = blockIdx.x * T_;
  const int tid = threadIdx.x, wave = tid >> 5, lane = tid & 31;
  const int fn = lane & 15, fhs = lane >> 4;

  if (wave == 0) {
    tdm_load_1d(0u, oF + (size_t)row0 * D_, (T_ * D_ * 2) / 8);
    __builtin_amdgcn_s_wait_tensorcnt(0);
  }
  __syncthreads();

  const int mt = wave;  // wave owns rows 16w..16w+15 for GEMM and LN
#pragma unroll
  for (int nt = 0; nt < 8; ++nt) {
    v8f c = {};
#pragma unroll
    for (int ks = 0; ks < 4; ++ks) {
      v16h a = frag_A(o_s + mt * 16 * D_ + ks * 32, D_);
      v16h b = frag_B_packed(WoP + ((nt * 4 + ks) << 9));
      c = wmma16(a, b, c);
    }
#pragma unroll
    for (int r = 0; r < 8; ++r) {
      const int m = r + 8 * fhs;
      const int rr = mt * 16 + m, cc = nt * 16 + fn;
      const float val = c[r] + x[(size_t)(row0 + rr) * D_ + cc] + bo[cc];
      x1_s[rr * D_ + cc] = val;
      out[(size_t)(row0 + rr) * D_ + cc] = val;
    }
  }

  // LN2 on own rows (same wave produced them)
  for (int r = 0; r < 16; ++r) {
    const int row = wave * 16 + r;
    const float4 v = *reinterpret_cast<const float4*>(x1_s + row * D_ + lane * 4);
    float s  = wave_sum(v.x + v.y + v.z + v.w);
    float s2 = wave_sum(v.x * v.x + v.y * v.y + v.z * v.z + v.w * v.w);
    const float mu = s * (1.0f / D_);
    const float rs = rsqrtf(s2 * (1.0f / D_) - mu * mu + EPS_);
    const float4 gg = *reinterpret_cast<const float4*>(g2 + lane * 4);
    const float4 bb = *reinterpret_cast<const float4*>(be2 + lane * 4);
    half_t* o = h2F + (size_t)(row0 + row) * D_ + lane * 4;
    o[0] = (half_t)((v.x - mu) * rs * gg.x + bb.x);
    o[1] = (half_t)((v.y - mu) * rs * gg.y + bb.y);
    o[2] = (half_t)((v.z - mu) * rs * gg.z + bb.z);
    o[3] = (half_t)((v.w - mu) * rs * gg.w + bb.w);
  }
}

// ---------------- K4: r = relu(h2 @ W1 + b1) -> f16 ------------------------

__global__ __launch_bounds__(256) void k_ffn1(
    const half_t* __restrict__ h2F, const half_t* __restrict__ W1P,
    const float* __restrict__ b1, half_t* __restrict__ rF) {
  half_t* h_s = (half_t*)smem_raw;  // 128x128 f16
  const int row0 = blockIdx.x * T_;
  const int tid = threadIdx.x, wave = tid >> 5, lane = tid & 31;
  const int fn = lane & 15, fhs = lane >> 4;
  if (wave == 0) {
    tdm_load_1d(0u, h2F + (size_t)row0 * D_, (T_ * D_ * 2) / 8);
    __builtin_amdgcn_s_wait_tensorcnt(0);
  }
  __syncthreads();

  const int mt = wave;
  for (int nt = 0; nt < 32; ++nt) {
    if (nt + 1 < 32)  // prefetch next weight tile (global_prefetch_b8)
      __builtin_prefetch(W1P + (((nt + 1) * 4) << 9), 0, 0);
    v8f c = {};
#pragma unroll
    for (int ks = 0; ks < 4; ++ks) {
      v16h a = frag_A(h_s + mt * 16 * D_ + ks * 32, D_);
      v16h b = frag_B_packed(W1P + ((nt * 4 + ks) << 9));
      c = wmma16(a, b, c);
    }
#pragma unroll
    for (int r = 0; r < 8; ++r) {
      const int m = r + 8 * fhs;
      const int cc = nt * 16 + fn;
      const float val = fmaxf(c[r] + b1[cc], 0.0f);
      rF[(size_t)(row0 + mt * 16 + m) * DFF + cc] = (half_t)val;
    }
  }
}

// ---------------- K5: d_out += r @ W2 + b2 ---------------------------------

__global__ __launch_bounds__(256) void k_ffn2(
    const half_t* __restrict__ rF, const half_t* __restrict__ W2P,
    const float* __restrict__ b2, float* __restrict__ out) {
  half_t* r_s = (half_t*)smem_raw;  // 128x512 f16 (128KB)
  const int row0 = blockIdx.x * T_;
  const int tid = threadIdx.x, wave = tid >> 5, lane = tid & 31;
  const int fn = lane & 15, fhs = lane >> 4;
  if (wave == 0) {
    tdm_load_1d(0u, rF + (size_t)row0 * DFF, (T_ * DFF * 2) / 8);
    __builtin_amdgcn_s_wait_tensorcnt(0);
  }
  __syncthreads();

  const int mt = wave;
#pragma unroll
  for (int nt = 0; nt < 8; ++nt) {
    if (nt + 1 < 8)  // prefetch next weight tile column
      __builtin_prefetch(W2P + (((nt + 1) * 16) << 9), 0, 0);
    v8f c = {};
#pragma unroll
    for (int ks = 0; ks < 16; ++ks) {
      v16h a = frag_A(r_s + mt * 16 * DFF + ks * 32, DFF);
      v16h b = frag_B_packed(W2P + ((nt * 16 + ks) << 9));
      c = wmma16(a, b, c);
    }
#pragma unroll
    for (int r = 0; r < 8; ++r) {
      const int m = r + 8 * fhs;
      const int rr = mt * 16 + m, cc = nt * 16 + fn;
      const size_t gi = (size_t)(row0 + rr) * D_ + cc;
      out[gi] = out[gi] + c[r] + b2[cc];
    }
  }
}

// ---------------- host launcher --------------------------------------------

extern "C" void kernel_launch(void* const* d_in, const int* in_sizes, int n_in,
                              void* d_out, int out_size, void* d_ws, size_t ws_size,
                              hipStream_t stream) {
  (void)in_sizes; (void)n_in; (void)out_size; (void)ws_size;
  const float* x    = (const float*)d_in[0];
  const float* Wq   = (const float*)d_in[1];
  const float* Wk   = (const float*)d_in[2];
  const float* Wv   = (const float*)d_in[3];
  const float* Wo   = (const float*)d_in[4];
  const float* bo   = (const float*)d_in[5];
  const float* W1   = (const float*)d_in[6];
  const float* b1   = (const float*)d_in[7];
  const float* W2   = (const float*)d_in[8];
  const float* b2   = (const float*)d_in[9];
  const float* g1   = (const float*)d_in[10];
  const float* be1  = (const float*)d_in[11];
  const float* g2   = (const float*)d_in[12];
  const float* be2  = (const float*)d_in[13];
  float* out = (float*)d_out;

  char* ws = (char*)d_ws;
  size_t off = 0;
  half_t* hF    = (half_t*)(ws + off); off += (size_t)B_ * T_ * D_ * 2;    // 16.8MB
  half_t* oF    = (half_t*)(ws + off); off += (size_t)B_ * T_ * D_ * 2;    // 16.8MB
  half_t* h2F   = (half_t*)(ws + off); off += (size_t)B_ * T_ * D_ * 2;    // 16.8MB
  half_t* rF    = (half_t*)(ws + off); off += (size_t)B_ * T_ * DFF * 2;   // 67.1MB
  half_t* WqkvP = (half_t*)(ws + off); off += (size_t)H_ * 3 * D_ * DHP * 2;
  half_t* WoP   = (half_t*)(ws + off); off += (size_t)D_ * D_ * 2;
  half_t* W1P   = (half_t*)(ws + off); off += (size_t)D_ * DFF * 2;
  half_t* W2P   = (half_t*)(ws + off); off += (size_t)DFF * D_ * 2;

  // allow >64KB dynamic LDS (gfx1250 WGP has 320KB); harmless if redundant
  (void)hipFuncSetAttribute((const void*)k_attn,
      hipFuncAttributeMaxDynamicSharedMemorySize, 122880);
  (void)hipFuncSetAttribute((const void*)k_proj_ln2,
      hipFuncAttributeMaxDynamicSharedMemorySize, 98304);
  (void)hipFuncSetAttribute((const void*)k_ffn2,
      hipFuncAttributeMaxDynamicSharedMemorySize, 131072);

  k_pack<<<864, 256, 0, stream>>>(Wq, Wk, Wv, Wo, W1, W2, WqkvP, WoP, W1P, W2P);
  k_ln1<<<(B_ * T_) / 8, 256, 0, stream>>>(x, g1, be1, hF);
  k_attn<<<B_ * H_, 256, 122880, stream>>>(hF, WqkvP, oF);
  k_proj_ln2<<<(B_ * T_) / T_, 256, 98304, stream>>>(oF, WoP, x, bo, g2, be2, out, h2F);
  k_ffn1<<<(B_ * T_) / T_, 256, 32768, stream>>>(h2F, W1P, b1, rF);
  k_ffn2<<<(B_ * T_) / T_, 256, 131072, stream>>>(rF, W2P, b2, out);
}
